// SigTKAN_42786464203330
// MI455X (gfx1250) — matrix operand order repaired and checked by reading, not compiled, not run on hardware
//
#include <hip/hip_runtime.h>

#define B_  128
#define T_  2048
#define F_  32
#define U_  64
#define D_  96
#define SIGN_ 1056            // F + F*F
#define ZS  104               // LDS row stride (halves); 208B = 16B-aligned rows
#define CH_ 128               // signature tile (timesteps per tile)
#define NT_ 16                // ceil((T-1)/CH) tiles

typedef __attribute__((ext_vector_type(16))) __bf16 v16bf;
typedef __attribute__((ext_vector_type(8)))  float  v8f;
typedef __attribute__((ext_vector_type(4)))  float  v4f;

#define LOG2E_  1.4426950408889634f
#define LOG2E2_ 2.8853900817779268f

// branch-free activations on the serial chain: native trans ops only
__device__ __forceinline__ float fast_sigmoid(float z) {
  return __builtin_amdgcn_rcpf(1.0f + __builtin_amdgcn_exp2f(-LOG2E_ * z));
}
__device__ __forceinline__ float fast_tanh(float z) {
#if __has_builtin(__builtin_amdgcn_tanhf)
  return __builtin_amdgcn_tanhf(z);                       // v_tanh_f32
#else
  return 1.0f - 2.0f * __builtin_amdgcn_rcpf(1.0f + __builtin_amdgcn_exp2f(LOG2E2_ * z));
#endif
}

// Workgroup barrier with ONLY the LDS-visibility wait (no loadcnt/storecnt
// fence, unlike __syncthreads) so register-prefetched global loads and
// fire-and-forget output stores stay in flight across the barrier.
__device__ __forceinline__ void lds_barrier() {
  asm volatile("s_wait_dscnt 0x0\n\t"
               "s_barrier_signal -1\n\t"
               "s_barrier_wait -1" ::: "memory");
}

// ---------------------------------------------------------------------------
// Phase 1a: signature. KEY: S1prev[t] = weighted[t] - weighted[0] in closed
// form, so lvl2 is a fully parallel reduction (no serial scan):
//   lvl2[i][j] = sum_t ((w_t - w_0)[i] + 0.5*dX_t[i]) * dX_t[j]
// One block (1024 thr) per batch; 16 double-buffered LDS tiles of 129 rows;
// tile n+1 is register-prefetched while tile n is reduced. 1 barrier/tile.
// ---------------------------------------------------------------------------
__global__ __launch_bounds__(1024) void sig_kernel(const float* __restrict__ x,
                                                   const float* __restrict__ tw,
                                                   float* __restrict__ sig) {
  __shared__ float rows[2][(CH_ + 1) * F_];
  const int b = blockIdx.x, tid = threadIdx.x;
  const int i = tid >> 5, j = tid & 31;
  const size_t xb = (size_t)b * T_ * F_;

  // stage tile 0 (rows 0..128, contiguous span -> fully coalesced)
#pragma unroll
  for (int q = 0; q < 5; ++q) {
    const int e = tid + q * 1024;
    if (e < (CH_ + 1) * F_) rows[0][e] = tw[e >> 5] * x[xb + e];
  }
  const float w0i = tw[0] * x[xb + i];
  __syncthreads();

  float acc = 0.0f;
  for (int tl = 0; tl < NT_; ++tl) {
    const int cur = tl & 1, nxt = cur ^ 1;
    const int t0 = tl * CH_;
    const int ns = min(CH_, (T_ - 1) - t0);

    // prefetch tile tl+1 into registers (overlaps with reduction below)
    float pre[5];
    int cntN = 0, t0N = (tl + 1) * CH_;
    if (tl + 1 < NT_) {
      cntN = (min(CH_, (T_ - 1) - t0N) + 1) * F_;
#pragma unroll
      for (int q = 0; q < 5; ++q) {
        const int e = tid + q * 1024;
        if (e < cntN) pre[q] = tw[t0N + (e >> 5)] * x[xb + (size_t)t0N * F_ + e];
      }
    }

    // reduce current tile
    const float* rb = rows[cur];
    float wti = rb[i], wtj = rb[j];
    for (int s = 0; s < ns; ++s) {
      const float w1i = rb[(s + 1) * F_ + i];
      const float w1j = rb[(s + 1) * F_ + j];
      const float dxi = w1i - wti, dxj = w1j - wtj;
      acc = fmaf((wti - w0i) + 0.5f * dxi, dxj, acc);
      wti = w1i; wtj = w1j;
    }

    // commit prefetched tile to the other LDS buffer
    if (tl + 1 < NT_) {
#pragma unroll
      for (int q = 0; q < 5; ++q) {
        const int e = tid + q * 1024;
        if (e < cntN) rows[nxt][e] = pre[q];
      }
    }
    __syncthreads();
  }

  sig[(size_t)b * SIGN_ + F_ + i * F_ + j] = acc;               // lvl2 flat
  if (tid < F_)                                                  // S1 closed form
    sig[(size_t)b * SIGN_ + tid] =
        tw[T_ - 1] * x[xb + (size_t)(T_ - 1) * F_ + tid] - tw[0] * x[xb + tid];
}

// ---------------------------------------------------------------------------
// Phase 1b: w = softmax(relu(sig@W1+b1)@W2+b2), one block (64 thr) per batch
// ---------------------------------------------------------------------------
__global__ __launch_bounds__(64) void gate_kernel(const float* __restrict__ sig,
                                                  const float* __restrict__ W1,
                                                  const float* __restrict__ b1,
                                                  const float* __restrict__ W2,
                                                  const float* __restrict__ b2,
                                                  float* __restrict__ wgate) {
  __shared__ float s[SIGN_];
  __shared__ float hbuf[U_];
  __shared__ float ybuf[U_];
  __shared__ float ebuf[U_];
  const int b = blockIdx.x, u = threadIdx.x;
  for (int k = u; k < SIGN_; k += U_) s[k] = sig[(size_t)b * SIGN_ + k];
  __syncthreads();
  float acc = b1[u];
  for (int k = 0; k < SIGN_; ++k) acc = fmaf(s[k], W1[k * U_ + u], acc);
  hbuf[u] = fmaxf(acc, 0.0f);
  __syncthreads();
  float y = b2[u];
  for (int v = 0; v < U_; ++v) y = fmaf(hbuf[v], W2[v * U_ + u], y);
  ybuf[u] = y;
  __syncthreads();
  float m = -3.0e38f;
  for (int v = 0; v < U_; ++v) m = fmaxf(m, ybuf[v]);
  const float e = __expf(y - m);
  ebuf[u] = e;
  __syncthreads();
  float ssum = 0.0f;
  for (int v = 0; v < U_; ++v) ssum += ebuf[v];
  wgate[(size_t)b * U_ + u] = e / ssum;
}

// ---------------------------------------------------------------------------
// Phase 2: recurrent scan. 8 blocks x 128 threads; block = 16 batch rows,
// wave w = output columns [16w,16w+16). Weights live in VGPRs as WMMA
// B-fragments; z tile (16x96 bf16) lives in LDS; 9 WMMAs per wave per step.
// x is register-prefetched at distance 2; barriers wait on DScnt only.
// ---------------------------------------------------------------------------
__global__ __launch_bounds__(128) void scan_kernel(
    const float* __restrict__ x,     const float* __restrict__ hinit,
    const float* __restrict__ Wi,    const float* __restrict__ bi,
    const float* __restrict__ Wo,    const float* __restrict__ bo,
    const float* __restrict__ Wg,    const float* __restrict__ bg,
    const float* __restrict__ wgate, float* __restrict__ out) {
  __shared__ alignas(16) __bf16 zbuf[16 * ZS];   // cols 0..31 = x_t, 32..95 = h

  const int b0   = blockIdx.x * 16;
  const int tid  = threadIdx.x;
  const int wave = tid >> 5, lane = tid & 31;
  const int lmod = lane & 15, lhi = lane >> 4;
  const int ncol = wave * 16 + lmod;

  // --- pack weight B-fragments (contiguous k order, same as A: lane n=lane%16 holds
  //     K = 32*kt + 16*(lane>=16) + 0..15, two halves per VGPR) ---
  const float* Wm[3] = {Wi, Wo, Wg};
  v16bf bf_[3][3];
#pragma unroll
  for (int g = 0; g < 3; ++g)
#pragma unroll
    for (int kt = 0; kt < 3; ++kt) {
      union { v16bf v; __bf16 e[16]; } bu;
#pragma unroll
      for (int kk = 0; kk < 16; ++kk)
        bu.e[kk] = (__bf16)Wm[g][(kt * 32 + lhi * 16 + kk) * U_ + ncol];
      bf_[g][kt] = bu.v;
    }
  const float bias_i = bi[ncol], bias_o = bo[ncol], bias_g = bg[ncol];
  float wf[8];
#pragma unroll
  for (int r = 0; r < 8; ++r)
    wf[r] = wgate[(size_t)(b0 + r + 8 * lhi) * U_ + ncol];

  // --- h0 = x[:,0,:] @ hidden_init into z h-region ---
  for (int e = tid; e < 16 * U_; e += 128) {
    const int m = e >> 6, n = e & 63;
    float a = 0.0f;
#pragma unroll
    for (int k = 0; k < F_; ++k)
      a = fmaf(x[((size_t)(b0 + m) * T_) * F_ + k], hinit[k * U_ + n], a);
    zbuf[m * ZS + F_ + n] = (__bf16)a;
  }

  // --- x staging: thread owns 4 consecutive floats (row xm, cols xk..xk+3),
  //     prefetch pipeline depth 2 ---
  const int xm = tid >> 3, xk = (tid & 7) * 4;
  const float* xrow = x + (size_t)(b0 + xm) * T_ * F_ + xk;
  v4f xc = *(const v4f*)(xrow);                            // t
  v4f xn = *(const v4f*)(xrow + F_);                       // t+1

  for (int t = 0; t < T_; ++t) {
    // store x_t tile (4 bf16 packed -> one ds_store_b64 per thread)
    union { __bf16 e[4]; unsigned long long u; } pk;
#pragma unroll
    for (int cc = 0; cc < 4; ++cc) pk.e[cc] = (__bf16)xc[cc];
    *reinterpret_cast<unsigned long long*>(&zbuf[xm * ZS + xk]) = pk.u;

    // prefetch x_{t+2}: waited two iterations from now
    v4f xn2 = xn;
    if (t + 2 < T_) xn2 = *(const v4f*)(xrow + (size_t)(t + 2) * F_);

    lds_barrier();   // x_t + h_t visible; prefetch/out-stores stay in flight

    // bias-initialized accumulators (all 8 C elements are column ncol)
    v8f ai, ao, ag;
#pragma unroll
    for (int r = 0; r < 8; ++r) { ai[r] = bias_i; ao[r] = bias_o; ag[r] = bias_g; }

#pragma unroll
    for (int kt = 0; kt < 3; ++kt) {
      // A fragment (contiguous k order, matching B): row lane%16,
      // K = 32*kt + 16*(lane>=16) + 0..15 -> two 16-byte LDS chunks
      union { v16bf v; v4f f4[2]; } au;
      const v4f* p = reinterpret_cast<const v4f*>(&zbuf[lmod * ZS + kt * 32 + lhi * 16]);
      au.f4[0] = p[0];
      au.f4[1] = p[1];
      ai = __builtin_amdgcn_wmma_f32_16x16x32_bf16(false, au.v, false, bf_[0][kt],
                                                   (short)0, ai, false, false);
      ao = __builtin_amdgcn_wmma_f32_16x16x32_bf16(false, au.v, false, bf_[1][kt],
                                                   (short)0, ao, false, false);
      ag = __builtin_amdgcn_wmma_f32_16x16x32_bf16(false, au.v, false, bf_[2][kt],
                                                   (short)0, ag, false, false);
    }

    // epilogue: C layout -> elem r is (row r+8*lhi, col lane%16); branch-free
    float res[8];
#pragma unroll
    for (int r = 0; r < 8; ++r) {
      const float iv = fast_sigmoid(ai[r]);
      const float ov = fast_sigmoid(ao[r]);
      const float gv = fast_tanh(ag[r]);
      const float v  = ov * fast_tanh(iv * gv) * wf[r];
      res[r] = v;
      __builtin_nontemporal_store(
          v, &out[((size_t)(b0 + r + 8 * lhi) * T_ + t) * U_ + ncol]);
    }

    lds_barrier();   // all A-fragment reads done before h update

#pragma unroll
    for (int r = 0; r < 8; ++r)
      zbuf[(r + 8 * lhi) * ZS + F_ + ncol] = (__bf16)res[r];

    xc = xn; xn = xn2;             // rotate prefetch pipeline
  }
}

// ---------------------------------------------------------------------------
extern "C" void kernel_launch(void* const* d_in, const int* in_sizes, int n_in,
                              void* d_out, int out_size, void* d_ws, size_t ws_size,
                              hipStream_t stream) {
  const float* x     = (const float*)d_in[0];
  const float* tw    = (const float*)d_in[1];
  const float* hinit = (const float*)d_in[2];
  const float* Wi    = (const float*)d_in[3];
  const float* bi    = (const float*)d_in[4];
  const float* Wo    = (const float*)d_in[5];
  const float* bo    = (const float*)d_in[6];
  const float* Wg    = (const float*)d_in[7];
  const float* bg    = (const float*)d_in[8];
  const float* W1    = (const float*)d_in[9];
  const float* b1    = (const float*)d_in[10];
  const float* W2    = (const float*)d_in[11];
  const float* b2    = (const float*)d_in[12];
  float* out = (float*)d_out;

  float* sig   = (float*)d_ws;              // B * 1056
  float* wgate = sig + (size_t)B_ * SIGN_;  // B * 64

  sig_kernel <<<B_, 1024, 0, stream>>>(x, tw, sig);
  gate_kernel<<<B_, U_,   0, stream>>>(sig, W1, b1, W2, b2, wgate);
  scan_kernel<<<B_ / 16, 128, 0, stream>>>(x, hinit, Wi, bi, Wo, bo, Wg, bg,
                                           wgate, out);
}
